// SivaRgr_class_17583596110014
// MI455X (gfx1250) — matrix operand order repaired and checked by reading, not compiled, not run on hardware
//
#include <hip/hip_runtime.h>
#include <hip/hip_bf16.h>

typedef __attribute__((ext_vector_type(16))) _Float16 v16h;
typedef __attribute__((ext_vector_type(8)))  _Float16 v8h;
typedef __attribute__((ext_vector_type(8)))  float    v8f;
typedef __attribute__((ext_vector_type(4)))  float    v4f;

#define TPB 128
#define A16S 40   // padded stride (halves) for f16 FFT-input arrays  [n1*A16S + n2]
#define B16S 40   // padded stride (halves) for f16 inter-stage arrays [n1 + B16S*k2]
#define C32S 36   // padded stride (floats) for f32 FFT-output arrays  [k1 + C32S*k2]

__device__ __forceinline__ v8f wmma_f16(v16h a, v16h b, v8f c) {
  // (neg_a, A, neg_b, B, c_mod, C, reuse_a, reuse_b); f16 WMMA has no A/B neg in ISA -> false
  return __builtin_amdgcn_wmma_f32_16x16x32_f16(false, a, false, b, (short)0, c, false, false);
}

// natural index n -> f16 input array index ; natural k -> f32 output array index
__device__ __forceinline__ int a16i(int n) { return (n & 31) * A16S + (n >> 5); }
__device__ __forceinline__ int c32i(int k) { return (k >> 5) + C32S * (k & 31); }

// 1024-pt complex FFT as four-step 32x32: X = F * ((xmat * F) .* T)
// f16 input (Ar16,Ai16) in fragment-friendly layout -> f32 output (Cr,Ci).
// DFT fragments/twiddles precomputed per thread, parked in LDS (pure b128 reloads).
// 'inv' is a literal at each call site, so slot selection constant-folds.
// Block = 128 threads = 4 waves; wave w owns output tile (tm,tn) = (w>>1, w&1).
__device__ __forceinline__ void fft1024(
    const _Float16* __restrict__ Ar16, const _Float16* __restrict__ Ai16,
    _Float16* __restrict__ Br16, _Float16* __restrict__ Bi16,
    float* __restrict__ Cr, float* __restrict__ Ci,
    const _Float16* __restrict__ fragH, const float* __restrict__ twF,
    int inv, float scale)
{
  const int tid  = threadIdx.x;
  const int lane = tid & 31;
  const int wid  = tid >> 5;
  const int tm   = wid >> 1;
  const int tn   = wid & 1;
  const int mrow = lane & 15;
  const int hi   = lane >> 4;

  const v16h* fp = (const v16h*)fragH;
  const v8f*  tp = (const v8f*)twF;
  // forward (inv=0): bp = -sin, bn = +sin, wi = -sin;  inverse (inv=1): swapped
  const v16h bc = fp[0 * TPB + tid];
  const v16h bp = fp[(inv ? 1 : 2) * TPB + tid];
  const v16h bn = fp[(inv ? 2 : 1) * TPB + tid];
  const v16h ac = fp[3 * TPB + tid];
  const v16h ap = fp[(inv ? 4 : 5) * TPB + tid];
  const v16h an = fp[(inv ? 5 : 4) * TPB + tid];
  const v8f  wc = tp[0 * TPB + tid];
  const v8f  wi = tp[(inv ? 1 : 2) * TPB + tid];

  __syncthreads();   // inputs ready

  // ---- stage 1: Y[n1][k2] = sum_n2 X[n1][n2] * F[n2][k2]; twiddle; -> B16
  // A-layout: elem e<8 -> K = e + hi*8 ; e>=8 -> K = e+8 + hi*8  (contiguous groups of 8)
  const int abase = (tm * 16 + mrow) * A16S + hi * 8;
  const v8h xr0 = *(const v8h*)(Ar16 + abase);
  const v8h xr1 = *(const v8h*)(Ar16 + abase + 16);
  const v8h xi0 = *(const v8h*)(Ai16 + abase);
  const v8h xi1 = *(const v8h*)(Ai16 + abase + 16);
  const v16h xr = __builtin_shufflevector(xr0, xr1, 0,1,2,3,4,5,6,7,8,9,10,11,12,13,14,15);
  const v16h xi = __builtin_shufflevector(xi0, xi1, 0,1,2,3,4,5,6,7,8,9,10,11,12,13,14,15);

  v8f yr = {}, yi = {};
  yr = wmma_f16(xr, bc, yr);   // Re += Xr*cos
  yr = wmma_f16(xi, bn, yr);   // Re -= Xi*(sgn*sin)
  yi = wmma_f16(xr, bp, yi);   // Im += Xr*(sgn*sin)
  yi = wmma_f16(xi, bc, yi);   // Im += Xi*cos

  {
    const int k2    = tn * 16 + mrow;
    const int bbase = tm * 16 + hi * 8 + B16S * k2;   // n1-contiguous in r
    v8h brv, biv;
    #pragma unroll
    for (int r = 0; r < 8; ++r) {
      const float a = yr[r], bb = yi[r];
      brv[r] = (_Float16)(a * wc[r] - bb * wi[r]);
      biv[r] = (_Float16)(a * wi[r] + bb * wc[r]);
    }
    *(v8h*)(Br16 + bbase) = brv;
    *(v8h*)(Bi16 + bbase) = biv;
  }
  __syncthreads();

  // ---- stage 2: Out[k1][k2] = sum_n1 F[k1][n1] * Z[n1][k2]; -> C (f32)
  // B-layout: elem e -> K = e + hi*16 (contiguous 16)
  const int zbase = B16S * (tn * 16 + mrow) + hi * 16;
  const v8h zr0 = *(const v8h*)(Br16 + zbase);
  const v8h zr1 = *(const v8h*)(Br16 + zbase + 8);
  const v8h zi0 = *(const v8h*)(Bi16 + zbase);
  const v8h zi1 = *(const v8h*)(Bi16 + zbase + 8);
  const v16h zr = __builtin_shufflevector(zr0, zr1, 0,1,2,3,4,5,6,7,8,9,10,11,12,13,14,15);
  const v16h zi = __builtin_shufflevector(zi0, zi1, 0,1,2,3,4,5,6,7,8,9,10,11,12,13,14,15);

  v8f outr = {}, outi = {};
  outr = wmma_f16(ac, zr, outr);
  outr = wmma_f16(an, zi, outr);
  outi = wmma_f16(ac, zi, outi);
  outi = wmma_f16(ap, zr, outi);

  {
    const int k2    = tn * 16 + mrow;
    const int cbase = tm * 16 + hi * 8 + C32S * k2;   // k1-contiguous in r
    v4f r0, r1, i0, i1;
    #pragma unroll
    for (int j = 0; j < 4; ++j) {
      r0[j] = scale * outr[j];
      r1[j] = scale * outr[j + 4];
      i0[j] = scale * outi[j];
      i1[j] = scale * outi[j + 4];
    }
    *(v4f*)(Cr + cbase)     = r0;
    *(v4f*)(Cr + cbase + 4) = r1;
    *(v4f*)(Ci + cbase)     = i0;
    *(v4f*)(Ci + cbase + 4) = i1;
  }
  __syncthreads();   // outputs visible
}

__global__ __launch_bounds__(TPB)
void siva_spectral_kernel(const float* __restrict__ x,
                          const float* __restrict__ nu_p,
                          const float* __restrict__ dtp_p,
                          const float* __restrict__ cr_p,
                          const int*   __restrict__ nT_p,
                          const int*   __restrict__ nS_p,
                          float* __restrict__ out,
                          int TC)
{
  __shared__ alignas(16) _Float16 Ar16[32 * A16S], Ai16[32 * A16S];  // FFT input (f16)
  __shared__ alignas(16) _Float16 Br16[32 * B16S], Bi16[32 * B16S];  // inter-stage (f16)
  __shared__ alignas(16) float    Cr[32 * C32S],   Ci[32 * C32S];    // FFT output (f32)
  __shared__ float uu[1024];             // current u(t), f32 across all substeps
  __shared__ float qinv[520];
  __shared__ alignas(32) _Float16 fragH[6 * TPB * 16];  // per-thread WMMA operand frags
  __shared__ alignas(32) float    twF[3 * TPB * 8];     // per-thread twiddle vectors

  const int b   = blockIdx.x;
  const int tid = threadIdx.x;

  const int lane = tid & 31;
  const int wid  = tid >> 5;
  const int tm   = wid >> 1;
  const int tn   = wid & 1;
  const int mrow = lane & 15;
  const int hi   = lane >> 4;

  const int   nTimes = nT_p[0];
  const int   nsub   = nS_p[0];
  const float nu  = 1.0f / (1.0f + expf(-nu_p[0]))  * (0.5f - 0.01f)  + 0.01f;
  const float dto = 1.0f / (1.0f + expf(-dtp_p[0])) * (0.1f - 0.001f) + 0.001f;
  const float cRt = 1.0f / (1.0f + expf(-cr_p[0]))  * (2.0f - 0.5f)   + 0.5f;
  const float dt  = dto / (float)nsub;

  const float TWO_PI = 6.2831853071795864769f;

  // ---- precompute per-thread WMMA fragments once; park in LDS ----
  #pragma unroll
  for (int e = 0; e < 16; ++e) {   // stage-1 B operand: F32[n2][k2]
    const int bK = ((e < 8) ? e : e + 8) + hi * 8;
    const int bN = tn * 16 + mrow;
    float sv, cv;
    __sincosf((TWO_PI / 32.0f) * (float)((bK * bN) & 31), &sv, &cv);
    fragH[(0 * TPB + tid) * 16 + e] = (_Float16)cv;
    fragH[(1 * TPB + tid) * 16 + e] = (_Float16)sv;
    fragH[(2 * TPB + tid) * 16 + e] = (_Float16)(-sv);
  }
  #pragma unroll
  for (int e = 0; e < 16; ++e) {   // stage-2 A operand: F32[k1][n1]
    const int aK   = e + hi * 16;
    const int arow = tm * 16 + mrow;
    float sv, cv;
    __sincosf((TWO_PI / 32.0f) * (float)((arow * aK) & 31), &sv, &cv);
    fragH[(3 * TPB + tid) * 16 + e] = (_Float16)cv;
    fragH[(4 * TPB + tid) * 16 + e] = (_Float16)sv;
    fragH[(5 * TPB + tid) * 16 + e] = (_Float16)(-sv);
  }
  #pragma unroll
  for (int r = 0; r < 8; ++r) {    // twiddle W_1024^{n1*k2}
    const int n1 = tm * 16 + r + hi * 8;
    const int k2 = tn * 16 + mrow;
    float sv, cv;
    __sincosf((TWO_PI / 1024.0f) * (float)(n1 * k2), &sv, &cv);
    twF[(0 * TPB + tid) * 8 + r] = cv;
    twF[(1 * TPB + tid) * 8 + r] = sv;
    twF[(2 * TPB + tid) * 8 + r] = -sv;
  }

  for (int k = tid; k <= 512; k += TPB) {
    const float fk = (float)k;
    qinv[k] = 1.0f / (1.0f - dt * cRt * (fk - nu * fk * fk));
  }
  for (int n = tid; n < 1024; n += TPB) {
    const int a = a16i(n);
    Ar16[a] = (_Float16)x[b * 1024 + n];
    Ai16[a] = (_Float16)0.0f;
  }

  #define FFT_FWD() fft1024(Ar16, Ai16, Br16, Bi16, Cr, Ci, fragH, twF, 0, 1.0f)
  #define FFT_INV() fft1024(Ar16, Ai16, Br16, Bi16, Cr, Ci, fragH, twF, 1, 1.0f / 1024.0f)

  // ---- d_to_u: u = irfft(i*k*rfft(d)) ----
  FFT_FWD();
  for (int k = tid; k <= 512; k += TPB) {
    const float fk = (float)k;
    const float hr = -fk * Ci[c32i(k)];
    const float hm =  fk * Cr[c32i(k)];
    const int a = a16i(k);
    Ar16[a] = (_Float16)hr; Ai16[a] = (_Float16)hm;
    if (k > 0 && k < 512) {
      const int am = a16i(1024 - k);
      Ar16[am] = (_Float16)hr; Ai16[am] = (_Float16)(-hm);
    }
  }
  FFT_INV();
  for (int n = tid; n < 1024; n += TPB) {
    const float un = Cr[c32i(n)];
    const int a = a16i(n);
    uu[n] = un; Ar16[a] = (_Float16)un; Ai16[a] = (_Float16)(0.5f * un * un);  // z = u + i*0.5u^2
  }

  for (int t = 0; t < nTimes; ++t) {
    for (int s = 0; s < nsub; ++s) {
      // one packed forward FFT gives rfft(u) and rfft(0.5u^2)
      FFT_FWD();
      for (int k = tid; k <= 512; k += TPB) {
        const int nk = (1024 - k) & 1023;
        const float zr = Cr[c32i(k)],  zm = Ci[c32i(k)];
        const float wr = Cr[c32i(nk)], wm = Ci[c32i(nk)];
        const float Ur = 0.5f * (zr + wr), Um = 0.5f * (zm - wm);  // uhat
        const float Vr = 0.5f * (zm + wm), Vm = 0.5f * (wr - zr);  // u2hat
        const float fk = (float)k, qi = qinv[k];
        const float nr = (Ur + dt * fk * Vm) * qi;                 // (U - dt*i*k*V)/q
        const float nm = (Um - dt * fk * Vr) * qi;
        const int a = a16i(k);
        Ar16[a] = (_Float16)nr; Ai16[a] = (_Float16)nm;
        if (k > 0 && k < 512) {
          const int am = a16i(1024 - k);
          Ar16[am] = (_Float16)nr; Ai16[am] = (_Float16)(-nm);
        }
      }
      FFT_INV();
      for (int n = tid; n < 1024; n += TPB) {
        const float un = Cr[c32i(n)];
        const int a = a16i(n);
        uu[n] = un; Ar16[a] = (_Float16)un; Ai16[a] = (_Float16)(0.5f * un * un);
      }
    }

    // ---- u_to_d: dsol = irfft(-i*rfft(u)/k), DC -> 0 ----
    for (int n = tid; n < 1024; n += TPB) {
      const int a = a16i(n);
      Ar16[a] = (_Float16)uu[n]; Ai16[a] = (_Float16)0.0f;
    }
    FFT_FWD();
    for (int k = tid; k <= 512; k += TPB) {
      float gr = 0.0f, gm = 0.0f;
      if (k > 0) {
        const float inv = 1.0f / (float)k;
        gr =  Ci[c32i(k)] * inv;
        gm = -Cr[c32i(k)] * inv;
      }
      const int a = a16i(k);
      Ar16[a] = (_Float16)gr; Ai16[a] = (_Float16)gm;
      if (k > 0 && k < 512) {
        const int am = a16i(1024 - k);
        Ar16[am] = (_Float16)gr; Ai16[am] = (_Float16)(-gm);
      }
    }
    FFT_INV();
    for (int n = tid; n < 1024; n += TPB) {
      out[b * (1024 * TC) + n * TC + t] = Cr[c32i(n)];   // [B, N, T*C]
    }
    // restore packed state for next outer block of substeps
    for (int n = tid; n < 1024; n += TPB) {
      const float un = uu[n];
      const int a = a16i(n);
      Ar16[a] = (_Float16)un; Ai16[a] = (_Float16)(0.5f * un * un);
    }
  }
  #undef FFT_FWD
  #undef FFT_INV
}

extern "C" void kernel_launch(void* const* d_in, const int* in_sizes, int n_in,
                              void* d_out, int out_size, void* d_ws, size_t ws_size,
                              hipStream_t stream) {
  (void)d_ws; (void)ws_size; (void)n_in;
  const float* x    = (const float*)d_in[0];
  const float* nu_p = (const float*)d_in[1];
  const float* dtp  = (const float*)d_in[2];
  const float* crp  = (const float*)d_in[3];
  const int*   nT   = (const int*)d_in[4];
  const int*   nS   = (const int*)d_in[5];
  float* out = (float*)d_out;

  const int B  = in_sizes[0] / 1024;        // 128
  const int TC = out_size / in_sizes[0];    // nTimes*C = 2

  hipLaunchKernelGGL(siva_spectral_kernel, dim3(B), dim3(TPB), 0, stream,
                     x, nu_p, dtp, crp, nT, nS, out, TC);
}
